// GraphTransformer_78434692759842
// MI455X (gfx1250) — matrix operand order repaired and checked by reading, not compiled, not run on hardware
//
#include <hip/hip_runtime.h>
#include <cstdint>

#define BS 4
#define NQ 256
#define NK 256
#define DX 256
#define DE 64
#define DY 128
#define HH 8
#define DF 32

typedef __attribute__((ext_vector_type(16))) __bf16 v16bf;
typedef __attribute__((ext_vector_type(8)))  float  v8f;
typedef __attribute__((ext_vector_type(4)))  unsigned int v4u;
typedef __attribute__((ext_vector_type(8)))  int v8i;
typedef __attribute__((ext_vector_type(4)))  int v4i;

// ---------------------------------------------------------------------------
// helpers
// ---------------------------------------------------------------------------
__device__ __forceinline__ unsigned short f2bf(float x) {
  union { float f; unsigned u; } c; c.f = x;
  unsigned r = c.u + 0x7FFFu + ((c.u >> 16) & 1u);   // round-to-nearest-even
  return (unsigned short)(r >> 16);
}

__device__ __forceinline__ v8f wmma_bf16(v16bf a, v16bf b, v8f c) {
  // D = A(16x32 bf16) * B(32x16 bf16) + C(16x16 f32)
  return __builtin_amdgcn_wmma_f32_16x16x32_bf16(
      /*neg_a=*/false, a, /*neg_b=*/false, b,
      /*c_mod=*/(short)0, c, /*reuse_a=*/false, /*reuse_b=*/false);
}

// generic LDS pointer -> LDS byte offset (flat LDS addr = addr[31:0], ISA 10.2)
__device__ __forceinline__ unsigned lds_addr_of(const void* p) {
  return (unsigned)(unsigned long long)p;
}

// -------- Tensor Data Mover: 1-D tile (nelem f32, contiguous) global -> LDS --
#if __has_builtin(__builtin_amdgcn_tensor_load_to_lds)
#define TDM_WAIT(n) __builtin_amdgcn_s_wait_tensorcnt((short)(n))
__device__ __forceinline__ void tdm_load_1d(unsigned lds_byte, const void* gaddr,
                                            unsigned nelem) {
  unsigned long long ga = (unsigned long long)gaddr;
  // D# group 0: count=1 | lds_addr | global_addr[56:0] | type=2
  v4u g0 = { 1u, lds_byte,
             (unsigned)(ga & 0xffffffffu),
             (unsigned)((ga >> 32) & 0x01ffffffu) | (2u << 30) };
  // D# group 1: data_size=4B, tensor_dim0=nelem, tensor_dim1=1,
  //             tile_dim0=nelem, tile_dim1=1, dim0_stride=nelem
  v8i g1;
  g1[0] = (int)(2u << 16);
  g1[1] = (int)((nelem & 0xffffu) << 16);
  g1[2] = (int)((nelem >> 16) & 0xffffu) | (1 << 16);
  g1[3] = (int)((nelem & 0xffffu) << 16);
  g1[4] = 1;
  g1[5] = (int)nelem;
  g1[6] = 0;
  g1[7] = 0;
  v4i gz = {0, 0, 0, 0};
#if defined(__clang_major__) && (__clang_major__ >= 23)
  v8i gz8 = {0, 0, 0, 0, 0, 0, 0, 0};
  __builtin_amdgcn_tensor_load_to_lds(g0, g1, gz, gz, gz8, 0);
#else
  __builtin_amdgcn_tensor_load_to_lds(g0, g1, gz, gz, 0);
#endif
}
#else
#define TDM_WAIT(n) asm volatile("s_wait_asynccnt %0" :: "i"(n) : "memory")
__device__ __forceinline__ void tdm_load_1d(unsigned lds_byte, const void* gaddr,
                                            unsigned nelem) {
  int lane = threadIdx.x & 31;
  const char* g = (const char*)gaddr;
  for (unsigned i = (unsigned)lane * 16u; i < nelem * 4u; i += 32u * 16u) {
    unsigned l = lds_byte + i;
    const char* p = g + i;
    asm volatile("global_load_async_to_lds_b128 %0, %1, off"
                 :: "v"(l), "v"(p) : "memory");
  }
}
#endif

// A-fragment from global f32 row-major [.., ld]; base points at (row0, k0).
__device__ __forceinline__ v16bf afrag_g(const float* base, int ld, int lane) {
  int half = lane >> 4, m = lane & 15;
  const float* p0 = base + (size_t)m * ld + half * 8;
  const float* p1 = p0 + 16;
  union { v16bf v; unsigned short u[16]; } r;
#pragma unroll
  for (int j = 0; j < 8; ++j) { r.u[j] = f2bf(p0[j]); r.u[8 + j] = f2bf(p1[j]); }
  return r.v;
}

// B-fragment: B[k][n] = W[n][k], W row-major [N][ld].
__device__ __forceinline__ v16bf bfrag_g(const float* W, int n0, int ld, int kk, int lane) {
  int half = lane >> 4, n = n0 + (lane & 15);
  const float* p = W + (size_t)n * ld + kk + half * 8;
  union { v16bf v; unsigned short u[16]; } r;
#pragma unroll
  for (int j = 0; j < 8; ++j) { r.u[j] = f2bf(p[j]); r.u[8 + j] = f2bf(p[16 + j]); }
  return r.v;
}

// A-fragment from LDS bf16 row-major [.., ld] (ushort-stored).
__device__ __forceinline__ v16bf afrag_lds(const unsigned short* S, int m0, int ld,
                                           int kk, int lane) {
  int half = lane >> 4, m = m0 + (lane & 15);
  const unsigned short* p0 = S + m * ld + kk + half * 8;
  union { v16bf v; unsigned short u[16]; } r;
#pragma unroll
  for (int j = 0; j < 8; ++j) { r.u[j] = p0[j]; r.u[8 + j] = p0[16 + j]; }
  return r.v;
}

// ---------------------------------------------------------------------------
// 1) C[M,N] = X[M,K] @ W[N,K]^T + bias   (one 16x16 tile per wave)
// ---------------------------------------------------------------------------
__global__ __launch_bounds__(32) void k_gemm_xwT_bias(
    const float* __restrict__ X, const float* __restrict__ W,
    const float* __restrict__ bias, float* __restrict__ C,
    int M, int N, int K) {
  int lane = threadIdx.x & 31;
  int tn = blockIdx.x * 16;
  int tm = blockIdx.y * 16;
  v8f acc = {};
  for (int kk = 0; kk < K; kk += 32) {
    v16bf a = afrag_g(X + (size_t)tm * K + kk, K, lane);
    v16bf b = bfrag_g(W, tn, K, kk, lane);
    acc = wmma_bf16(a, b, acc);
  }
  int half = lane >> 4;
  int col = tn + (lane & 15);
  float bv = bias[col];
#pragma unroll
  for (int r = 0; r < 8; ++r)
    C[(size_t)(tm + half * 8 + r) * N + col] = acc[r] + bv;
}

// ---------------------------------------------------------------------------
// 2) per-batch y-derived modulation vectors: ye1, ye2, yx1, yx2 (256) + y2 (128)
// ---------------------------------------------------------------------------
__global__ __launch_bounds__(256) void k_yvec(
    const float* __restrict__ y,
    const float* yeaW, const float* yeab, const float* yemW, const float* yemb,
    const float* yxaW, const float* yxab, const float* yxmW, const float* yxmb,
    const float* ypW, const float* ypb, float* __restrict__ out) {
  __shared__ float yr[DY];
  int b = blockIdx.x, t = threadIdx.x;
  if (t < DY) yr[t] = y[b * DY + t];
  __syncthreads();
  float* ob = out + b * 1152;
  float s1 = yeab[t], s2 = yemb[t], s3 = yxab[t], s4 = yxmb[t];
  for (int k = 0; k < DY; ++k) {
    float yk = yr[k];
    s1 += yk * yeaW[t * DY + k];
    s2 += yk * yemW[t * DY + k];
    s3 += yk * yxaW[t * DY + k];
    s4 += yk * yxmW[t * DY + k];
  }
  ob[t] = s1; ob[256 + t] = s2; ob[512 + t] = s3; ob[768 + t] = s4;
  if (t < DY) {
    float s5 = ypb[t];
    for (int k = 0; k < DY; ++k) s5 += yr[k] * ypW[t * DY + k];
    ob[1024 + t] = s5;
  }
}

// ---------------------------------------------------------------------------
// 3) masked stats over E: per-(b,q) partials (sum, sumsq, min, max) per channel
// ---------------------------------------------------------------------------
__global__ __launch_bounds__(256) void k_estats_partial(
    const float* __restrict__ E, const unsigned char* __restrict__ mq,
    const unsigned char* __restrict__ mk, float* __restrict__ part) {
  __shared__ float rs[256], rq[256], rmn[256], rmx[256];
  int bq = blockIdx.x; int b = bq / NQ;
  int t = threadIdx.x; int ks = t >> 6;
  float s = 0.f, sq = 0.f, mn = 1e30f, mx = -1e30f;
  if (mq[bq]) {
    const float* Eb = E + (size_t)bq * NK * DE;
    int ch = t & 63;
    for (int k = ks; k < NK; k += 4) {
      if (mk[b * NK + k]) {
        float x = Eb[k * DE + ch];
        s += x; sq += x * x; mn = fminf(mn, x); mx = fmaxf(mx, x);
      }
    }
  }
  rs[t] = s; rq[t] = sq; rmn[t] = mn; rmx[t] = mx;
  __syncthreads();
  if (t < 64) {
    for (int j = 64; j < 256; j += 64) {
      s += rs[t + j]; sq += rq[t + j];
      mn = fminf(mn, rmn[t + j]); mx = fmaxf(mx, rmx[t + j]);
    }
    float* p = part + ((size_t)bq * 64 + t) * 4;
    p[0] = s; p[1] = sq; p[2] = mn; p[3] = mx;
  }
}

__global__ __launch_bounds__(64) void k_estats_final(
    const float* __restrict__ part, const unsigned char* __restrict__ mq,
    const unsigned char* __restrict__ mk, float* __restrict__ se) {
  int b = blockIdx.x, ch = threadIdx.x;
  float s = 0.f, sq = 0.f, mn = 1e30f, mx = -1e30f;
  for (int q = 0; q < NQ; ++q) {
    const float* p = part + (((size_t)(b * NQ + q)) * 64 + ch) * 4;
    s += p[0]; sq += p[1]; mn = fminf(mn, p[2]); mx = fmaxf(mx, p[3]);
  }
  int cq = 0, ck = 0;
  for (int i = 0; i < NQ; ++i) cq += mq[b * NQ + i] ? 1 : 0;
  for (int i = 0; i < NK; ++i) ck += mk[b * NK + i] ? 1 : 0;
  float cnt = (float)cq * (float)ck;
  float denom = (cnt == 0.f) ? 1.f : cnt;
  float mean = s / denom;
  // reference zeroes unmasked entries *before* subtracting the mean:
  float var = (sq - 2.f * mean * s + (float)(NQ * NK) * mean * mean) / denom + 1e-10f;
  float sd = (cnt < 2.f) ? 0.f : sqrtf(var);
  if (mn > 0.5e30f) mn = 0.f;
  if (mx < -0.5e30f) mx = 0.f;
  float* o = se + b * 256;
  o[ch] = mean; o[64 + ch] = sd; o[128 + ch] = mn; o[192 + ch] = mx;
}

// ---------------------------------------------------------------------------
// 4) masked stats over Xq / Xk  (N=256 rows, 256 channels)
// ---------------------------------------------------------------------------
__global__ __launch_bounds__(256) void k_xstats(
    const float* __restrict__ X, const unsigned char* __restrict__ m,
    float* __restrict__ out) {
  int b = blockIdx.x, ch = threadIdx.x;
  float s = 0.f, sq = 0.f, mn = 1e30f, mx = -1e30f; int cnt = 0;
  const float* Xb = X + (size_t)b * NQ * DX;
  for (int i = 0; i < NQ; ++i) {
    if (m[b * NQ + i]) {
      float x = Xb[i * DX + ch];
      s += x; sq += x * x; mn = fminf(mn, x); mx = fmaxf(mx, x); ++cnt;
    }
  }
  float c = (float)cnt, denom = (c == 0.f) ? 1.f : c;
  float mean = s / denom;
  float var = (sq - 2.f * mean * s + (float)NQ * mean * mean) / denom + 1e-10f;
  float sd = (c < 2.f) ? 0.f : sqrtf(var);
  if (mn > 0.5e30f) mn = 0.f;
  if (mx < -0.5e30f) mx = 0.f;
  float* o = out + b * 4 * DX;
  o[ch] = mean; o[DX + ch] = sd; o[2 * DX + ch] = mn; o[3 * DX + ch] = mx;
}

// ---------------------------------------------------------------------------
// 5) final y MLP: cat -> relu(lin) -> lin
// ---------------------------------------------------------------------------
__global__ __launch_bounds__(128) void k_yfinal(
    const float* __restrict__ yv, const float* __restrict__ sxq,
    const float* __restrict__ sxk, const float* __restrict__ se,
    const float* rxqW, const float* rxqb, const float* rxkW, const float* rxkb,
    const float* reW, const float* reb, const float* yo1W, const float* yo1b,
    const float* yo2W, const float* yo2b, float* __restrict__ outY) {
  __shared__ float cat[4 * DY];
  __shared__ float hid[DY];
  int b = blockIdx.x, t = threadIdx.x;
  cat[t] = yv[b * 1152 + 1024 + t];
  float a = rxqb[t], c2 = rxkb[t], e = reb[t];
  const float* xq = sxq + b * 1024;
  const float* xk = sxk + b * 1024;
  const float* sev = se + b * 256;
  for (int k = 0; k < 1024; ++k) {
    a += xq[k] * rxqW[t * 1024 + k];
    c2 += xk[k] * rxkW[t * 1024 + k];
  }
  for (int k = 0; k < 256; ++k) e += sev[k] * reW[t * 256 + k];
  cat[DY + t] = a; cat[2 * DY + t] = c2; cat[3 * DY + t] = e;
  __syncthreads();
  float h = yo1b[t];
  for (int k = 0; k < 4 * DY; ++k) h += cat[k] * yo1W[t * 4 * DY + k];
  hid[t] = fmaxf(h, 0.f);
  __syncthreads();
  float o = yo2b[t];
  for (int k = 0; k < DY; ++k) o += hid[k] * yo2W[t * DY + k];
  outY[b * DY + t] = o;
}

// ---------------------------------------------------------------------------
// 6) fused per-(b,q) kernel.  Dynamic-LDS partition (123136 bytes):
//    [0      ) sE   bf16  E slice           32 KB   (persistent)
//    [32768  ) sA   bf16  handoff tile       8 KB   (persistent)
//    [40960  ) sW   f32   head scores        8 KB   (persistent)
//    [49152  ) 8x f32[256] vectors + eob   ~8.4 KB  (persistent)
//    [57600  ) sK   f32   K double-buffer   32 KB   (overlaid below)
//    [57600  ) sEf32 f32  TDM E staging     64 KB   (dead after convert)
// ---------------------------------------------------------------------------
#define SM_SE    0
#define SM_SA    32768
#define SM_SW    40960
#define SM_VEC   49152
#define SM_K     57600
#define SM_EF32  57600
#define SM_TOTAL 123136

__global__ __launch_bounds__(256) void k_fused(
    const float* __restrict__ E, const float* __restrict__ Qg,
    const float* __restrict__ Kg, const float* __restrict__ Vg,
    const float* __restrict__ eaW, const float* __restrict__ eab,
    const float* __restrict__ emW, const float* __restrict__ emb,
    const float* __restrict__ eoW, const float* __restrict__ eob,
    const float* __restrict__ xoW, const float* __restrict__ xob,
    const float* __restrict__ yv,
    const unsigned char* __restrict__ mq, const unsigned char* __restrict__ mk,
    float* __restrict__ outX, float* __restrict__ outE) {
  extern __shared__ __align__(16) char smem[];
  unsigned short* sE = (unsigned short*)(smem + SM_SE);
  unsigned short* sA = (unsigned short*)(smem + SM_SA);
  float* sW    = (float*)(smem + SM_SW);
  float* sQ    = (float*)(smem + SM_VEC);
  float* sYe1  = sQ + 256;
  float* sYe2  = sYe1 + 256;
  float* sYx1  = sYe2 + 256;
  float* sYx2  = sYx1 + 256;
  float* sEab  = sYx2 + 256;
  float* sEmb  = sEab + 256;
  float* sX    = sEmb + 256;
  float* sEob  = sX + 256;
  float* sK    = (float*)(smem + SM_K);
  float* sEf32 = (float*)(smem + SM_EF32);

  int bq = blockIdx.x; int b = bq / NQ;
  int t = threadIdx.x; int lane = t & 31; int wave = t >> 5;
  int half = lane >> 4;

  const float* Eb = E + (size_t)bq * NK * DE;
  const float* Kb = Kg + (size_t)b * NK * DX;

  // kick off the TDM copy of the E slice first, overlap setup behind it
  if (wave == 0) tdm_load_1d(lds_addr_of(sEf32), Eb, NK * DE);

  // per-block vectors + score zeroing (overlaps the DMA)
  sQ[t]   = Qg[(size_t)bq * DX + t];
  sYe1[t] = yv[b * 1152 + t];
  sYe2[t] = yv[b * 1152 + 256 + t];
  sYx1[t] = yv[b * 1152 + 512 + t];
  sYx2[t] = yv[b * 1152 + 768 + t];
  sEab[t] = eab[t]; sEmb[t] = emb[t];
  if (t < DE) sEob[t] = eob[t];
  for (int i = t; i < NK * HH; i += 256) sW[i] = 0.f;
  __builtin_prefetch(xoW + (size_t)t * DX, 0, 1);   // warm L2 for tail GEMV

  // B-fragments live in registers (reused by all 16 k-tiles); also behind DMA
  v16bf bea[2][2], bem[2][2];
#pragma unroll
  for (int ti = 0; ti < 2; ++ti) {
    int nt = wave * 2 + ti;
#pragma unroll
    for (int kssub = 0; kssub < 2; ++kssub) {
      bea[ti][kssub] = bfrag_g(eaW, nt * 16, DE, kssub * 32, lane);
      bem[ti][kssub] = bfrag_g(emW, nt * 16, DE, kssub * 32, lane);
    }
  }
  v16bf beo[8];
  if (wave < 4) {
#pragma unroll
    for (int kssub = 0; kssub < 8; ++kssub)
      beo[kssub] = bfrag_g(eoW, wave * 16, DX, kssub * 32, lane);
  }

  if (wave == 0) { TDM_WAIT(0); }
  __syncthreads();

  // convert staged E slice to bf16
  for (int i = t; i < NK * DE; i += 256) sE[i] = f2bf(sEf32[i]);
  __syncthreads();

  // prologue: start DMA of K tile 0 (overlays the now-dead staging region)
  if (wave == 0) tdm_load_1d(lds_addr_of(sK), Kb, 16 * DX);

  int mqv = mq[bq] ? 1 : 0;
  const float rs32 = 0.17677669529663689f;  // 1/sqrt(DF)

  for (int kt = 0; kt < 16; ++kt) {
    // double-buffered K tiles: issue next, wait for current
    if (wave == 0) {
      if (kt < 15) {
        tdm_load_1d(lds_addr_of(sK + ((kt + 1) & 1) * (16 * DX)),
                    Kb + (size_t)(kt + 1) * 16 * DX, 16 * DX);
        TDM_WAIT(1);
      } else {
        TDM_WAIT(0);
      }
    }
    __syncthreads();
    const float* sKc = sK + (kt & 1) * (16 * DX);

    v16bf af0 = afrag_lds(sE, kt * 16, DE, 0, lane);
    v16bf af1 = afrag_lds(sE, kt * 16, DE, 32, lane);
#pragma unroll
    for (int ti = 0; ti < 2; ++ti) {
      int ncol = (wave * 2 + ti) * 16 + (lane & 15);
      v8f a1 = {}; v8f a2 = {};
      a1 = wmma_bf16(af0, bea[ti][0], a1);
      a1 = wmma_bf16(af1, bea[ti][1], a1);
      a2 = wmma_bf16(af0, bem[ti][0], a2);
      a2 = wmma_bf16(af1, bem[ti][1], a2);
      float e1b = sEab[ncol], e2b = sEmb[ncol];
      float qv = sQ[ncol] * rs32;
      float y1 = sYe1[ncol], y2 = sYe2[ncol];
      int h = ncol >> 5;
#pragma unroll
      for (int r = 0; r < 8; ++r) {
        int krow = kt * 16 + half * 8 + r;
        float kvv = sKc[(half * 8 + r) * DX + ncol];
        float Aval = (a1[r] + e1b) + ((a2[r] + e2b) + 1.f) * (qv * kvv);
        atomicAdd(&sW[krow * HH + h], Aval);                 // ds_add_f32
        sA[(half * 8 + r) * DX + ncol] = f2bf(y1 + (y2 + 1.f) * Aval);
      }
    }
    __syncthreads();
    if (wave < 4) {   // eo GEMM: [16 x 256] @ eoW^T -> [16 x 64]
      v8f acc = {};
#pragma unroll
      for (int kssub = 0; kssub < 8; ++kssub) {
        v16bf af = afrag_lds(sA, 0, DX, kssub * 32, lane);
        acc = wmma_bf16(af, beo[kssub], acc);
      }
      int ncol = wave * 16 + (lane & 15);
      float bb = sEob[ncol];
#pragma unroll
      for (int r = 0; r < 8; ++r) {
        int krow = kt * 16 + half * 8 + r;
        float msk = (mqv && mk[b * NK + krow]) ? 1.f : 0.f;
        outE[((size_t)bq * NK + krow) * DE + ncol] = (acc[r] + bb) * msk;
      }
    }
    __syncthreads();
  }

  // masked softmax over k, per head (mk[b,0] is guaranteed true)
  if (t < HH) {
    int h = t;
    float mmax = -1e30f;
    for (int k = 0; k < NK; ++k)
      if (mk[b * NK + k]) mmax = fmaxf(mmax, sW[k * HH + h]);
    float ssum = 0.f;
    for (int k = 0; k < NK; ++k) {
      float e = mk[b * NK + k] ? __expf(sW[k * HH + h] - mmax) : 0.f;
      sW[k * HH + h] = e; ssum += e;
    }
    float inv = 1.f / ssum;
    for (int k = 0; k < NK; ++k) sW[k * HH + h] *= inv;
  }
  __syncthreads();

  // attn @ V (V slice is L2-resident), y-modulation
  {
    int h = t >> 5;
    float acc = 0.f;
    for (int k = 0; k < NK; ++k)
      acc += sW[k * HH + h] * Vg[((size_t)b * NK + k) * DX + t];
    sX[t] = sYx1[t] + (sYx2[t] + 1.f) * acc;
  }
  __syncthreads();
  {
    float o = xob[t];
    for (int k = 0; k < DX; ++k) o += sX[k] * xoW[t * DX + k];
    outX[(size_t)bq * DX + t] = mqv ? o : 0.f;
  }
}

// ---------------------------------------------------------------------------
// launch
// ---------------------------------------------------------------------------
extern "C" void kernel_launch(void* const* d_in, const int* in_sizes, int n_in,
                              void* d_out, int out_size, void* d_ws, size_t ws_size,
                              hipStream_t stream) {
  (void)in_sizes; (void)n_in; (void)out_size; (void)ws_size;

  const float* Xq  = (const float*)d_in[0];
  const float* Xk  = (const float*)d_in[1];
  const float* E   = (const float*)d_in[2];
  const float* y   = (const float*)d_in[3];
  const float* qW  = (const float*)d_in[4];
  const float* qb  = (const float*)d_in[5];
  const float* kW  = (const float*)d_in[6];
  const float* kb  = (const float*)d_in[7];
  const float* vW  = (const float*)d_in[8];
  const float* vb  = (const float*)d_in[9];
  const float* eaW = (const float*)d_in[10];
  const float* eab = (const float*)d_in[11];
  const float* emW = (const float*)d_in[12];
  const float* emb = (const float*)d_in[13];
  const float* yeaW = (const float*)d_in[14];
  const float* yeab = (const float*)d_in[15];
  const float* yemW = (const float*)d_in[16];
  const float* yemb = (const float*)d_in[17];
  const float* yxaW = (const float*)d_in[18];
  const float* yxab = (const float*)d_in[19];
  const float* yxmW = (const float*)d_in[20];
  const float* yxmb = (const float*)d_in[21];
  const float* ypW  = (const float*)d_in[22];
  const float* ypb  = (const float*)d_in[23];
  const float* rxqW = (const float*)d_in[24];
  const float* rxqb = (const float*)d_in[25];
  const float* rxkW = (const float*)d_in[26];
  const float* rxkb = (const float*)d_in[27];
  const float* reW  = (const float*)d_in[28];
  const float* reb  = (const float*)d_in[29];
  const float* xoW  = (const float*)d_in[30];
  const float* xob  = (const float*)d_in[31];
  const float* eoW  = (const float*)d_in[32];
  const float* eob  = (const float*)d_in[33];
  const float* yo1W = (const float*)d_in[34];
  const float* yo1b = (const float*)d_in[35];
  const float* yo2W = (const float*)d_in[36];
  const float* yo2b = (const float*)d_in[37];
  const unsigned char* mq = (const unsigned char*)d_in[38];
  const unsigned char* mk = (const unsigned char*)d_in[39];

  float* ws = (float*)d_ws;
  float* Qg    = ws;                 // 1024*256
  float* Kg    = ws + 262144;        // 1024*256
  float* Vg    = ws + 524288;        // 1024*256
  float* yvv   = ws + 786432;        // 4*1152
  float* epart = ws + 791040;        // 1024*64*4
  float* se    = ws + 1053184;       // 4*256
  float* sxq   = ws + 1054208;       // 4*1024
  float* sxk   = ws + 1058304;       // 4*1024

  float* outX = (float*)d_out;                       // 4*256*256
  float* outE = outX + (size_t)BS * NQ * DX;         // 4*256*256*64
  float* outY = outE + (size_t)BS * NQ * NK * DE;    // 4*128

  // QKV projections (WMMA bf16)
  dim3 gg(DX / 16, (BS * NQ) / 16);
  k_gemm_xwT_bias<<<gg, 32, 0, stream>>>(Xq, qW, qb, Qg, BS * NQ, DX, DX);
  k_gemm_xwT_bias<<<gg, 32, 0, stream>>>(Xk, kW, kb, Kg, BS * NK, DX, DX);
  k_gemm_xwT_bias<<<gg, 32, 0, stream>>>(Xk, vW, vb, Vg, BS * NK, DX, DX);

  // y-derived vectors + stats path
  k_yvec<<<BS, 256, 0, stream>>>(y, yeaW, yeab, yemW, yemb, yxaW, yxab,
                                 yxmW, yxmb, ypW, ypb, yvv);
  k_estats_partial<<<BS * NQ, 256, 0, stream>>>(E, mq, mk, epart);
  k_estats_final<<<BS, 64, 0, stream>>>(epart, mq, mk, se);
  k_xstats<<<BS, 256, 0, stream>>>(Xq, mq, sxq);
  k_xstats<<<BS, 256, 0, stream>>>(Xk, mk, sxk);
  k_yfinal<<<BS, 128, 0, stream>>>(yvv, sxq, sxk, se, rxqW, rxqb, rxkW, rxkb,
                                   reW, reb, yo1W, yo1b, yo2W, yo2b, outY);

  // big fused kernel: newE + newX  (TDM-staged E and K tiles)
  k_fused<<<BS * NQ, 256, SM_TOTAL, stream>>>(E, Qg, Kg, Vg, eaW, eab, emW, emb,
                                              eoW, eob, xoW, xob, yvv, mq, mk,
                                              outX, outE);
}
